// SparseConv2D_1915555414386
// MI455X (gfx1250) — matrix operand order repaired and checked by reading, not compiled, not run on hardware
//
#include <hip/hip_runtime.h>

// ---------------- problem constants (from reference) ----------------
#define N_IMG 16
#define C_IN 128
#define H_IN 64
#define W_IN 64
#define O_CH 128
#define OHH 62
#define OWW 62
#define K_TOT 1152                 // C_IN * 3 * 3
#define PIX (OHH * OWW)            // 3844
#define M_TOT (N_IMG * PIX)        // 61504
#define NK (K_TOT / 32)            // 36 K-steps

// ---------------- tiling ----------------
#define MB 128                     // M tile per block
#define KB 32                      // K step (one bf16 WMMA depth)
#define LDA 48                     // padded LDS row length (bf16 elems) -> 96B rows, 16/32B aligned

typedef __bf16 bf16;
typedef bf16  v8bf  __attribute__((ext_vector_type(8)));
typedef bf16  v16bf __attribute__((ext_vector_type(16)));
typedef float v8f   __attribute__((ext_vector_type(8)));
typedef int   v4i   __attribute__((ext_vector_type(4)));
typedef float v4f   __attribute__((ext_vector_type(4)));

__device__ __forceinline__ bf16 bf_trunc(float f) {
    unsigned u = __builtin_bit_cast(unsigned, f);
    unsigned short h = (unsigned short)(u >> 16);
    return __builtin_bit_cast(bf16, h);
}
__device__ __forceinline__ float bf_hi_f32(float f) {
    unsigned u = __builtin_bit_cast(unsigned, f) & 0xffff0000u;
    return __builtin_bit_cast(float, u);
}

__global__ __launch_bounds__(256) void conv3x3_wmma_bf16x3(
    const float* __restrict__ x,      // [N,C,H,W]
    const float* __restrict__ w,      // [O, C*9]  (c-major, then kh, kw)
    const float* __restrict__ bias,   // [O]
    float* __restrict__ out)          // [N,O,62,62]
{
    // 64KB LDS:
    //   [    0..12288)  A hi  [MB][LDA] bf16
    //   [12288..24576)  A lo
    //   [24576..36864)  B hi  [O_CH][LDA] bf16 (o-major)
    //   [36864..49152)  B lo
    //   [49152..53760)  im2col x-offset LUT, int[K_TOT]
    //   epilogue: Cs float[128][128] overlays everything
    __shared__ __attribute__((aligned(128))) char smem[65536];
    bf16* Ah = (bf16*)(smem);
    bf16* Al = (bf16*)(smem + 12288);
    bf16* Bh = (bf16*)(smem + 24576);
    bf16* Bl = (bf16*)(smem + 36864);
    int*  lut = (int*)(smem + 49152);
    float* Cs = (float*)(smem);

    const int tid  = threadIdx.x;
    const int lane = tid & 31;
    const int wid  = tid >> 5;
    const int m_base = blockIdx.x * MB;

    // ---- one-time im2col offset LUT: off(k) = c*H*W + kh*W + kw ----
    for (int k = tid; k < K_TOT; k += 256) {
        int c  = k / 9;
        int r9 = k - 9 * c;
        int kh = r9 / 3;
        int kw = r9 - 3 * kh;
        lut[k] = c * (H_IN * W_IN) + kh * W_IN + kw;
    }

    // ---- staging coordinates: thread t stages row (t/2), 16 k-elems at (t&1)*16 ----
    const int sm = tid >> 1;
    const int sk = (tid & 1) * 16;

    // im2col decomposition of this thread's A row (loop-invariant)
    const int mg = m_base + sm;
    const bool mvalid = (mg < M_TOT);
    const int mc = mvalid ? mg : 0;
    const int n  = mc / PIX;
    const int rp = mc - n * PIX;
    const int oh = rp / OWW;
    const int ow = rp - oh * OWW;
    const float* xbase = x + (((size_t)n * C_IN) * H_IN + oh) * W_IN + ow; // x[n][0][oh][ow]
    const float* wrow  = w + (size_t)sm * K_TOT + sk;                       // weight[sm][sk + ...]

    // ---- wave tile: 8 waves in 4(M) x 2(O) grid; each wave 32x64 = 2x4 WMMA tiles ----
    const int wave_m = (wid & 3) * 32;
    const int wave_o = (wid >> 2) * 64;
    const int row16  = lane & 15;
    const int hi16   = lane >> 4;      // 0 or 1

    v8f acc[2][4];
#pragma unroll
    for (int i = 0; i < 2; ++i)
#pragma unroll
        for (int j = 0; j < 4; ++j) acc[i][j] = v8f{};

    __syncthreads();                   // LUT visible

    // ---- software pipeline registers: staged A gather + B rows for current step ----
    float ra[16];
    union { float e[16]; v4f v[4]; } rb;

    // prologue: load K-step 0
    {
        const v4i* l4 = (const v4i*)(lut + sk);
#pragma unroll
        for (int q = 0; q < 4; ++q) {
            v4i o4 = l4[q];
#pragma unroll
            for (int j = 0; j < 4; ++j)
                ra[q * 4 + j] = mvalid ? xbase[o4[j]] : 0.0f;
        }
        const v4f* w4 = (const v4f*)(wrow);
#pragma unroll
        for (int q = 0; q < 4; ++q) rb.v[q] = w4[q];
    }

    for (int kb = 0; kb < NK; ++kb) {
        __syncthreads();   // previous iteration's compute done reading LDS

        // ---------- stage current step: hi/lo split + vector LDS stores ----------
        {
            union { bf16 e[16]; v8bf v[2]; } hA, lA, hB, lB;
#pragma unroll
            for (int e = 0; e < 16; ++e) {
                float v  = ra[e];
                float vh = bf_hi_f32(v);
                hA.e[e] = bf_trunc(v);
                lA.e[e] = bf_trunc(v - vh);
                float b  = rb.e[e];
                float bh = bf_hi_f32(b);
                hB.e[e] = bf_trunc(b);
                lB.e[e] = bf_trunc(b - bh);
            }
            v8bf* pAh = (v8bf*)(Ah + sm * LDA + sk);
            v8bf* pAl = (v8bf*)(Al + sm * LDA + sk);
            v8bf* pBh = (v8bf*)(Bh + sm * LDA + sk);
            v8bf* pBl = (v8bf*)(Bl + sm * LDA + sk);
            pAh[0] = hA.v[0]; pAh[1] = hA.v[1];
            pAl[0] = lA.v[0]; pAl[1] = lA.v[1];
            pBh[0] = hB.v[0]; pBh[1] = hB.v[1];
            pBl[0] = lB.v[0]; pBl[1] = lB.v[1];
        }
        __syncthreads();

        // ---------- issue next step's global loads (latency hidden under WMMAs) ----------
        if (kb + 1 < NK) {
            const int kg = (kb + 1) * KB;
            const v4i* l4 = (const v4i*)(lut + kg + sk);
#pragma unroll
            for (int q = 0; q < 4; ++q) {
                v4i o4 = l4[q];
#pragma unroll
                for (int j = 0; j < 4; ++j)
                    ra[q * 4 + j] = mvalid ? xbase[o4[j]] : 0.0f;
            }
            const v4f* w4 = (const v4f*)(wrow + kg);
#pragma unroll
            for (int q = 0; q < 4; ++q) rb.v[q] = w4[q];
            if (kb + 2 < NK) __builtin_prefetch(wrow + kg + KB, 0, 1); // global_prefetch
        }

        // ---------- load fragments (A and B both in contiguous k = 16*half + s order) ----------
        v16bf afh[2], afl[2], bfh[4], bfl[4];
        const int k0 = hi16 * 16;      // A: K chunks {k0..k0+7} and {k0+8..k0+15}
#pragma unroll
        for (int tm = 0; tm < 2; ++tm) {
            int row = wave_m + tm * 16 + row16;
            v8bf a0 = *(const v8bf*)(Ah + row * LDA + k0);
            v8bf a1 = *(const v8bf*)(Ah + row * LDA + k0 + 8);
            afh[tm] = __builtin_shufflevector(a0, a1, 0,1,2,3,4,5,6,7,8,9,10,11,12,13,14,15);
            v8bf l0 = *(const v8bf*)(Al + row * LDA + k0);
            v8bf l1 = *(const v8bf*)(Al + row * LDA + k0 + 8);
            afl[tm] = __builtin_shufflevector(l0, l1, 0,1,2,3,4,5,6,7,8,9,10,11,12,13,14,15);
        }
        const int koff = hi16 * 16;    // B: lane holds column, 16 contiguous K
#pragma unroll
        for (int to = 0; to < 4; ++to) {
            int orow = wave_o + to * 16 + row16;
            bfh[to] = *(const v16bf*)(Bh + orow * LDA + koff);
            bfl[to] = *(const v16bf*)(Bl + orow * LDA + koff);
        }

        // ---------- 24 WMMAs per wave per K-step: hi*hi + hi*lo + lo*hi ----------
#pragma unroll
        for (int tm = 0; tm < 2; ++tm)
#pragma unroll
            for (int to = 0; to < 4; ++to) {
                acc[tm][to] = __builtin_amdgcn_wmma_f32_16x16x32_bf16(
                    false, afh[tm], false, bfh[to], (short)0, acc[tm][to], false, false);
                acc[tm][to] = __builtin_amdgcn_wmma_f32_16x16x32_bf16(
                    false, afh[tm], false, bfl[to], (short)0, acc[tm][to], false, false);
                acc[tm][to] = __builtin_amdgcn_wmma_f32_16x16x32_bf16(
                    false, afl[tm], false, bfh[to], (short)0, acc[tm][to], false, false);
            }
    }

    // ---------- epilogue: transpose via LDS, coalesced NCHW stores + bias ----------
    __syncthreads();
#pragma unroll
    for (int tm = 0; tm < 2; ++tm)
#pragma unroll
        for (int to = 0; to < 4; ++to) {
            int ocol  = wave_o + to * 16 + row16;          // C/D: lane = column (O)
            int mrow0 = wave_m + tm * 16 + hi16 * 8;       // VGPR v = row (M), +8 for upper half
#pragma unroll
            for (int v = 0; v < 8; ++v)
                Cs[ocol * MB + mrow0 + v] = acc[tm][to][v];
        }
    __syncthreads();

    {
        const int mlane = tid & 127;
        const int ohalf = tid >> 7;                        // 0/1
        const int mg2 = m_base + mlane;
        if (mg2 < M_TOT) {
            int n2 = mg2 / PIX;
            int r2 = mg2 - n2 * PIX;
            float* obase = out + (size_t)n2 * (O_CH * PIX) + r2;
#pragma unroll 8
            for (int i = 0; i < 64; ++i) {
                int o = ohalf + 2 * i;
                obase[(size_t)o * PIX] = Cs[o * MB + mlane] + bias[o];
            }
        }
    }
}

extern "C" void kernel_launch(void* const* d_in, const int* in_sizes, int n_in,
                              void* d_out, int out_size, void* d_ws, size_t ws_size,
                              hipStream_t stream) {
    const float* x    = (const float*)d_in[0];
    const float* wgt  = (const float*)d_in[1];
    const float* bias = (const float*)d_in[2];
    float* out        = (float*)d_out;

    const int grid = (M_TOT + MB - 1) / MB;   // 481 blocks of 256 threads (8 waves)
    conv3x3_wmma_bf16x3<<<grid, 256, 0, stream>>>(x, wgt, bias, out);
}
